// Attention_16071767621814
// MI455X (gfx1250) — matrix operand order repaired and checked by reading, not compiled, not run on hardware
//
#include <hip/hip_runtime.h>
#include <hip/hip_bf16.h>
#include <stdint.h>

// ---------------------------------------------------------------------------
// Problem constants (from reference)
// ---------------------------------------------------------------------------
#define BATCH 16
#define DIM   192
#define HEADS 8
#define HID   64
#define INNER 512          // HEADS*HID
#define HIMG  32
#define WIMG  32
#define LQ    1024         // 32*32
#define LK    256          // 16*16
#define EPSBN 1e-5f

typedef __bf16 bf16_t;
typedef __attribute__((ext_vector_type(16))) bf16_t v16bf;
typedef __attribute__((ext_vector_type(8)))  float  v8f;

union FragBf { uint32_t u[8]; uint4 q[2]; v16bf v; };

// float -> bf16 bits (round to nearest even)
__device__ __forceinline__ uint16_t f2bf(float f) {
    union { float f; uint32_t u; } in; in.f = f;
    uint32_t r = (in.u + 0x7FFFu + ((in.u >> 16) & 1u)) >> 16;
    return (uint16_t)r;
}

// ---------------------------------------------------------------------------
// WMMA fragment layouts (CDNA5 ISA §7.12.2, wave32)
//   A (16x32, k order used here): lanes 0-15: M=lane, V0..3 K=0..7 (pairs), V4..7 K=8..15;
//              lanes16-31: M=lane-16, K offsets +16.
//   B (32x16, same k order as A): lanes 0-15: N=lane, V0..7 K=0..15 (pairs); lanes16-31 K=16..31.
//   C (16x16): VGPR j, lanes 0-15 -> (M=j, N=lane); lanes16-31 -> (M=j+8).
// ---------------------------------------------------------------------------

// A fragment from row-major [m][k] bf16, lda elements (vectorizes to 2xb128).
__device__ __forceinline__ v16bf load_a_rm(const uint16_t* A, int lda, int k0, int lane) {
    FragBf f;
    const int g = (lane >> 4) & 1;
    const int m = lane & 15;
    const uint16_t* row = A + (size_t)m * lda + k0;
#pragma unroll
    for (int v = 0; v < 4; ++v) {
        int k = 2 * v + 16 * g;
        f.u[v] = *(const uint32_t*)(row + k);
    }
#pragma unroll
    for (int v = 4; v < 8; ++v) {
        int k = 2 * v + 16 * g;
        f.u[v] = *(const uint32_t*)(row + k);
    }
    return f.v;
}

// B fragment from "WF" fragment-tiled storage: per (n-tile nt, k-step ks),
// 512 bf16 laid out as [lane][16]: each lane reads 32 contiguous bytes.
__device__ __forceinline__ v16bf load_b_wf(const uint16_t* WF, int kSteps,
                                           int ks, int nt, int lane) {
    const uint16_t* p = WF + (((size_t)nt * kSteps + ks) * 32 + lane) * 16;
    FragBf f;
    const uint4* q = (const uint4*)p;      // 32B-aligned by construction
    f.q[0] = q[0];
    f.q[1] = q[1];
    return f.v;
}

__device__ __forceinline__ v8f wmma_bf16(v16bf a, v16bf b, v8f c) {
    return __builtin_amdgcn_wmma_f32_16x16x32_bf16(false, a, false, b, (short)0, c, false, false);
}

// ---------------------------------------------------------------------------
// Kernel 1: 1x1-conv weight (N,K) f32 row-major -> WF fragment-tiled bf16.
// WF index: (((nt*kSteps + ks)*32 + lane)*16 + 2*v + h)
//   with n = nt*16 + (lane&15), k = ks*32 + 2*v + 16*(lane>>4) + h.
// ---------------------------------------------------------------------------
__global__ void wtrans_wf_kernel(const float* __restrict__ W, uint16_t* __restrict__ WF,
                                 int N, int K) {
    int idx = blockIdx.x * blockDim.x + threadIdx.x;
    if (idx >= N * K) return;
    const int kSteps = K >> 5;
    int pos  = idx & 15;
    int lane = (idx >> 4) & 31;
    int ks   = (idx >> 9) % kSteps;
    int nt   = idx / (512 * kSteps);
    int v = pos >> 1, h = pos & 1;
    int n = nt * 16 + (lane & 15);
    int k = ks * 32 + 2 * v + 16 * (lane >> 4) + h;
    WF[idx] = f2bf(W[(size_t)n * K + k]);
}

// ---------------------------------------------------------------------------
// Kernel 2: depthwise 3x3 conv + BatchNorm(eval) -> token-major bf16 (B,L,DIM)
// ---------------------------------------------------------------------------
__global__ void dwbn_tok_kernel(const float* __restrict__ x,
                                const float* __restrict__ dw,
                                const float* __restrict__ bg,
                                const float* __restrict__ bb,
                                const float* __restrict__ bm,
                                const float* __restrict__ bv,
                                uint16_t* __restrict__ tok,
                                int stride, int OH, int OW) {
    const int c  = threadIdx.x;                    // 0..191
    const int bl = blockIdx.x;
    const int L  = OH * OW;
    const int b  = bl / L;
    const int l  = bl % L;
    const int oy = l / OW, ox = l % OW;

    float acc = 0.f;
#pragma unroll
    for (int ky = 0; ky < 3; ++ky) {
        int iy = oy * stride - 1 + ky;
        if (iy < 0 || iy >= HIMG) continue;
#pragma unroll
        for (int kx = 0; kx < 3; ++kx) {
            int ix = ox * stride - 1 + kx;
            if (ix < 0 || ix >= WIMG) continue;
            acc += x[(((size_t)b * DIM + c) * HIMG + iy) * WIMG + ix] * dw[c * 9 + ky * 3 + kx];
        }
    }
    float s = bg[c] * rsqrtf(bv[c] + EPSBN);
    float y = acc * s + (bb[c] - bm[c] * s);
    tok[((size_t)b * L + l) * DIM + c] = f2bf(y);
}

// ---------------------------------------------------------------------------
// Kernel 3: batched bf16 GEMM, one 16x16 tile per wave, B in WF layout.
//   C(MxN) = A(MxK, bf16 row-major, lda) * B(WF fragment-tiled, K x N)
//   mode 0: store bf16 row-major (ldc = N)
//   mode 1: store fp32 transposed (out[n][m], ld = M) + bias[n]  (final conv)
//   mode 2: store bf16 WF for a consumer reading C as B (consumer K = this M)
//   mode 3: store bf16 per-head TRANSPOSED WF: consumer reads C^T as B with
//           k = channel-within-head (HID), n = token (M). Used for attn K^T.
// grid: (N/64, M/16, batch), block 128 (4 waves; wave w takes n-tile w)
// ---------------------------------------------------------------------------
__global__ void gemm_bf16_kernel(const uint16_t* __restrict__ A,
                                 const uint16_t* __restrict__ B,
                                 void* __restrict__ Cout,
                                 const float* __restrict__ bias,
                                 int M, int N, int K, int lda,
                                 long aBatch, long cBatch, int mode) {
    const int wave = threadIdx.x >> 5;
    const int lane = threadIdx.x & 31;
    const int n0   = (blockIdx.x * 4 + wave) * 16;
    const int m0   = blockIdx.y * 16;
    const int bat  = blockIdx.z;
    if (n0 >= N) return;

    const uint16_t* Ab = A + (size_t)bat * aBatch + (size_t)m0 * lda;
    const int kSteps = K >> 5;
    const int nt = n0 >> 4;

    v8f acc = {};
    for (int ks = 0; ks < kSteps; ++ks) {
        v16bf a = load_a_rm(Ab, lda, ks * 32, lane);
        v16bf b = load_b_wf(B, kSteps, ks, nt, lane);
        acc = wmma_bf16(a, b, acc);
    }

    const int g   = (lane >> 4) & 1;
    const int col = lane & 15;
    if (mode == 0) {
        uint16_t* C = (uint16_t*)Cout + (size_t)bat * cBatch;
#pragma unroll
        for (int j = 0; j < 8; ++j) {
            int row = j + 8 * g;
            C[(size_t)(m0 + row) * N + n0 + col] = f2bf(acc[j]);
        }
    } else if (mode == 1) {
        float* C = (float*)Cout + (size_t)bat * cBatch;
        float bvv = bias ? bias[n0 + col] : 0.f;
#pragma unroll
        for (int j = 0; j < 8; ++j) {
            int row = j + 8 * g;
            C[(size_t)(n0 + col) * M + (m0 + row)] = acc[j] + bvv;
        }
    } else if (mode == 2) {
        // WF store: consumer sees this matrix as B(K'=M x N), kSteps' = M/32.
        uint16_t* C = (uint16_t*)Cout + (size_t)bat * cBatch;
        const int kSteps2 = M >> 5;
#pragma unroll
        for (int j = 0; j < 8; ++j) {
            int m  = m0 + j + 8 * g;               // row of C == k of consumer
            int ks2   = m >> 5;
            int kr    = m & 31;
            int lane2 = (kr & 16) + col;
            int v2    = (kr & 15) >> 1;
            int h2    = kr & 1;
            C[(((size_t)nt * kSteps2 + ks2) * 32 + lane2) * 16 + 2 * v2 + h2] = f2bf(acc[j]);
        }
    } else {
        // Transposed per-head WF store: consumer B(K'=HID x N'=M) per head.
        // k' = channel-within-head, n' = token row m. kSteps' = HID/32 = 2.
        uint16_t* C = (uint16_t*)Cout + (size_t)bat * cBatch;
        int c_  = n0 + col;
        int hh  = c_ >> 6;                         // head
        int d   = c_ & (HID - 1);                  // channel within head
        int ks2 = d >> 5;
        int kr  = d & 31;
        int kg  = kr & 16;
        int v2  = (kr & 15) >> 1;
        int h2  = kr & 1;
        uint16_t* Ch = C + (size_t)hh * (HID * LK);
#pragma unroll
        for (int j = 0; j < 8; ++j) {
            int m     = m0 + j + 8 * g;            // token == consumer n
            int nt2   = m >> 4;
            int lane2 = kg + (m & 15);
            Ch[(((size_t)nt2 * 2 + ks2) * 32 + lane2) * 16 + 2 * v2 + h2] = f2bf(acc[j]);
        }
    }
}

// ---------------------------------------------------------------------------
// Kernel 4: attention. One wave owns a 16-query strip of one (b,h):
//   S(16x256) = Qh * Kh^T / 8 -> softmax (VGPR-resident, shuffle reductions)
//   P -> LDS bf16 -> AV(16x64) = P * Vh.
// K^T and V are WF fragment-tiled, so all B loads are contiguous b128s.
// grid: B*HEADS*(LQ/64) = 2048 blocks, 128 threads (4 waves * 16 rows)
// ---------------------------------------------------------------------------
__global__ void attention_kernel(const uint16_t* __restrict__ Q,
                                 const uint16_t* __restrict__ Kwf,
                                 const uint16_t* __restrict__ Vwf,
                                 uint16_t* __restrict__ AV) {
    __shared__ uint16_t Pl[4][16 * LK];            // 4 waves * 8KB

    const int wave = threadIdx.x >> 5;
    const int lane = threadIdx.x & 31;
    const int mblk = blockIdx.x & 15;
    const int bh   = blockIdx.x >> 4;
    const int h    = bh & 7;
    const int b    = bh >> 3;
    const int m0   = mblk * 64 + wave * 16;

    const uint16_t* Qb = Q   + ((size_t)b * LQ + m0) * INNER + h * HID;
    const uint16_t* Kb = Kwf + (size_t)b * LK * INNER + (size_t)h * (HID * LK);
    const uint16_t* Vb = Vwf + (size_t)b * LK * INNER;     // WF, nt = h*4+dt

    // ---- S = Q K^T : 16 accumulator tiles cover all 256 keys -------------
    v16bf a0 = load_a_rm(Qb, INNER, 0,  lane);
    v16bf a1 = load_a_rm(Qb, INNER, 32, lane);

    v8f s[16];
#pragma unroll
    for (int kt = 0; kt < 16; ++kt) {
        v8f c = {};
        c = wmma_bf16(a0, load_b_wf(Kb, 2, 0, kt, lane), c);
        c = wmma_bf16(a1, load_b_wf(Kb, 2, 1, kt, lane), c);
        s[kt] = c;
    }

    // ---- softmax over 256 keys; row j lives in one 16-lane group ---------
    const float scale = 0.125f;                    // 1/sqrt(64)
    uint16_t* P = &Pl[wave][0];
#pragma unroll
    for (int j = 0; j < 8; ++j) {
        float mx = -3.4e38f;
#pragma unroll
        for (int kt = 0; kt < 16; ++kt) { s[kt][j] *= scale; mx = fmaxf(mx, s[kt][j]); }
#pragma unroll
        for (int msk = 8; msk >= 1; msk >>= 1) mx = fmaxf(mx, __shfl_xor(mx, msk, 32));
        float sum = 0.f;
#pragma unroll
        for (int kt = 0; kt < 16; ++kt) { float e = __expf(s[kt][j] - mx); s[kt][j] = e; sum += e; }
#pragma unroll
        for (int msk = 8; msk >= 1; msk >>= 1) sum += __shfl_xor(sum, msk, 32);
        float inv = 1.f / sum;
        int row = j + 8 * ((lane >> 4) & 1);
        int col = lane & 15;
#pragma unroll
        for (int kt = 0; kt < 16; ++kt)
            P[row * LK + kt * 16 + col] = f2bf(s[kt][j] * inv);
    }
    __syncthreads();

    // ---- AV = P * V ------------------------------------------------------
    v16bf pa[8];
#pragma unroll
    for (int ks = 0; ks < 8; ++ks) pa[ks] = load_a_rm(P, LK, ks * 32, lane);

    const int g   = (lane >> 4) & 1;
    const int col = lane & 15;
#pragma unroll
    for (int dt = 0; dt < 4; ++dt) {
        v8f acc = {};
#pragma unroll
        for (int ks = 0; ks < 8; ++ks) {
            v16bf bv = load_b_wf(Vb, LK / 32, ks, h * 4 + dt, lane);
            acc = wmma_bf16(pa[ks], bv, acc);
        }
#pragma unroll
        for (int j = 0; j < 8; ++j) {
            int row = j + 8 * g;
            AV[((size_t)b * LQ + m0 + row) * INNER + h * HID + dt * 16 + col] = f2bf(acc[j]);
        }
    }
}

// ---------------------------------------------------------------------------
// Host-side launch
// ---------------------------------------------------------------------------
extern "C" void kernel_launch(void* const* d_in, const int* in_sizes, int n_in,
                              void* d_out, int out_size, void* d_ws, size_t ws_size,
                              hipStream_t stream) {
    const float* x    = (const float*)d_in[0];
    const float* qdw  = (const float*)d_in[1];
    const float* qbg  = (const float*)d_in[2];
    const float* qbb  = (const float*)d_in[3];
    const float* qbm  = (const float*)d_in[4];
    const float* qbv  = (const float*)d_in[5];
    const float* qpw  = (const float*)d_in[6];
    const float* kdw  = (const float*)d_in[7];
    const float* kbg  = (const float*)d_in[8];
    const float* kbb  = (const float*)d_in[9];
    const float* kbm  = (const float*)d_in[10];
    const float* kbv  = (const float*)d_in[11];
    const float* kpw  = (const float*)d_in[12];
    const float* vdw  = (const float*)d_in[13];
    const float* vbg  = (const float*)d_in[14];
    const float* vbb  = (const float*)d_in[15];
    const float* vbm  = (const float*)d_in[16];
    const float* vbv  = (const float*)d_in[17];
    const float* vpw  = (const float*)d_in[18];
    const float* ow   = (const float*)d_in[19];
    const float* ob   = (const float*)d_in[20];
    float* out        = (float*)d_out;

    // Workspace carve-up (256B aligned)
    char* ws = (char*)d_ws;
    size_t off = 0;
    auto carve = [&](size_t bytes) {
        void* p = ws + off;
        off = (off + bytes + 255) & ~(size_t)255;
        return p;
    };
    uint16_t* Aq  = (uint16_t*)carve((size_t)BATCH * LQ * DIM * 2);
    uint16_t* Ak  = (uint16_t*)carve((size_t)BATCH * LK * DIM * 2);
    uint16_t* Av  = (uint16_t*)carve((size_t)BATCH * LK * DIM * 2);
    uint16_t* WqF = (uint16_t*)carve((size_t)DIM * INNER * 2);
    uint16_t* WkF = (uint16_t*)carve((size_t)DIM * INNER * 2);
    uint16_t* WvF = (uint16_t*)carve((size_t)DIM * INNER * 2);
    uint16_t* OwF = (uint16_t*)carve((size_t)INNER * DIM * 2);
    uint16_t* Qm  = (uint16_t*)carve((size_t)BATCH * LQ * INNER * 2);
    uint16_t* Km  = (uint16_t*)carve((size_t)BATCH * LK * INNER * 2);
    uint16_t* Vm  = (uint16_t*)carve((size_t)BATCH * LK * INNER * 2);
    uint16_t* AVt = (uint16_t*)carve((size_t)BATCH * LQ * INNER * 2);
    (void)ws_size;

    // 1) weights -> WF fragment-tiled bf16
    {
        int n = INNER * DIM;
        wtrans_wf_kernel<<<(n + 255) / 256, 256, 0, stream>>>(qpw, WqF, INNER, DIM);
        wtrans_wf_kernel<<<(n + 255) / 256, 256, 0, stream>>>(kpw, WkF, INNER, DIM);
        wtrans_wf_kernel<<<(n + 255) / 256, 256, 0, stream>>>(vpw, WvF, INNER, DIM);
        wtrans_wf_kernel<<<(n + 255) / 256, 256, 0, stream>>>(ow,  OwF, DIM, INNER);
    }

    // 2) dw-conv + BN -> bf16 tokens
    dwbn_tok_kernel<<<BATCH * LQ, DIM, 0, stream>>>(x, qdw, qbg, qbb, qbm, qbv, Aq, 1, 32, 32);
    dwbn_tok_kernel<<<BATCH * LK, DIM, 0, stream>>>(x, kdw, kbg, kbb, kbm, kbv, Ak, 2, 16, 16);
    dwbn_tok_kernel<<<BATCH * LK, DIM, 0, stream>>>(x, vdw, vbg, vbb, vbm, vbv, Av, 2, 16, 16);

    // 3) pointwise projections (WMMA GEMMs)
    gemm_bf16_kernel<<<dim3(INNER / 64, LQ / 16, BATCH), 128, 0, stream>>>(
        Aq, WqF, Qm, nullptr, LQ, INNER, DIM, DIM,
        (long)LQ * DIM, (long)LQ * INNER, 0);               // Q: bf16 row-major
    gemm_bf16_kernel<<<dim3(INNER / 64, LK / 16, BATCH), 128, 0, stream>>>(
        Ak, WkF, Km, nullptr, LK, INNER, DIM, DIM,
        (long)LK * DIM, (long)LK * INNER, 3);               // K: per-head WF^T
    gemm_bf16_kernel<<<dim3(INNER / 64, LK / 16, BATCH), 128, 0, stream>>>(
        Av, WvF, Vm, nullptr, LK, INNER, DIM, DIM,
        (long)LK * DIM, (long)LK * INNER, 2);               // V: WF layout

    // 4) attention (QK^T -> softmax -> AV), all WMMA, all-b128 B loads
    attention_kernel<<<BATCH * HEADS * (LQ / 64), 128, 0, stream>>>(Qm, Km, Vm, AVt);

    // 5) output 1x1 conv: fp32 NCHW store + bias, straight into d_out
    gemm_bf16_kernel<<<dim3(DIM / 64, LQ / 16, BATCH), 128, 0, stream>>>(
        AVt, OwF, out, ob, LQ, DIM, INNER, INNER,
        (long)LQ * INNER, (long)DIM * LQ, 1);
}
